// Attention_36249523978627
// MI455X (gfx1250) — matrix operand order repaired and checked by reading, not compiled, not run on hardware
//
#include <hip/hip_runtime.h>
#include <hip/hip_bf16.h>

#define B_   2048
#define T_   26
#define DIN_ 512
#define H_   512
#define E_   256
#define C_   97
#define S_   32

typedef __attribute__((ext_vector_type(16))) __bf16 v16bf;
typedef __attribute__((ext_vector_type(8)))  float  v8f;

// ---------- helpers ----------
__device__ __forceinline__ unsigned short f2bf(float f) {
    unsigned u = __float_as_uint(f);
    unsigned r = (u >> 16) & 1u;
    u += 0x7fffu + r;                       // round to nearest even
    return (unsigned short)(u >> 16);
}
__device__ __forceinline__ float bf2f(unsigned short b) {
    return __uint_as_float(((unsigned)b) << 16);
}
__device__ __forceinline__ float sigmoidf_(float x) {
    return 1.0f / (1.0f + __expf(-x));
}

// ---------- one-time prep kernels ----------
__global__ void cvt_bf16_kernel(const float* __restrict__ src,
                                unsigned short* __restrict__ dst, size_t n) {
    size_t i = (size_t)blockIdx.x * blockDim.x + threadIdx.x;
    if (i < n) dst[i] = f2bf(src[i]);
}

// Wcat[n][k], n in [0,2048), k in [0,1280): k<768 -> W_ih[n][k], else W_hh[n][k-768]
__global__ void prep_wcat_kernel(const float* __restrict__ W_ih,
                                 const float* __restrict__ W_hh,
                                 const float* __restrict__ b_ih,
                                 const float* __restrict__ b_hh,
                                 unsigned short* __restrict__ Wcat,
                                 float* __restrict__ bias) {
    size_t idx = (size_t)blockIdx.x * blockDim.x + threadIdx.x;
    const size_t total = (size_t)(4 * H_) * 1280;
    if (idx < total) {
        int n = (int)(idx / 1280);
        int k = (int)(idx % 1280);
        float v = (k < 768) ? W_ih[(size_t)n * 768 + k]
                            : W_hh[(size_t)n * 512 + (k - 768)];
        Wcat[idx] = f2bf(v);
    }
    if (idx < (size_t)(4 * H_)) bias[idx] = b_ih[idx] + b_hh[idx];
}

// Pad W_gen [97,512] -> [256,512] bf16, zero rows beyond 97
__global__ void prep_wgen_kernel(const float* __restrict__ Wg,
                                 unsigned short* __restrict__ Wp) {
    int idx = blockIdx.x * blockDim.x + threadIdx.x;   // 256*512
    if (idx >= 256 * 512) return;
    int n = idx >> 9, k = idx & 511;
    Wp[idx] = (n < C_) ? f2bf(Wg[(size_t)n * 512 + k]) : (unsigned short)0;
}

// zero x (h region matters for step 0) and cell state c
__global__ void zero_init_kernel(unsigned short* __restrict__ x,
                                 float* __restrict__ c) {
    size_t idx = (size_t)blockIdx.x * blockDim.x + threadIdx.x;
    if (idx < (size_t)B_ * 1280) x[idx] = 0;
    if (idx < (size_t)B_ * H_)   c[idx] = 0.0f;
}

// ---------- WMMA GEMM: C[M,N] = A[M,K](bf16,row) * W[N,K](bf16,row)^T + bias ----------
// Requires: M%64==0, N%256==0, K%32==0. Store-guarded at col<Nstore.
__global__ __launch_bounds__(256)
void gemm_bf16_nt_kernel(const unsigned short* __restrict__ A, int lda,
                         const unsigned short* __restrict__ W, int K,
                         void* __restrict__ Cout, int ldc,
                         const float* __restrict__ bias,
                         int Nstore, int out_bf16) {
    const int wave = threadIdx.x >> 5;
    const int lane = threadIdx.x & 31;
    const int wm = wave >> 2;               // 0..1
    const int wn = wave & 3;                // 0..3
    const int m0 = blockIdx.x * 64 + wm * 32;
    const int n0 = blockIdx.y * 256 + wn * 64;
    const int lr = lane & 15;
    const int lh = lane >> 4;               // half-wave: 0 or 1

    v8f acc[2][4];
#pragma unroll
    for (int i = 0; i < 2; ++i)
#pragma unroll
        for (int j = 0; j < 4; ++j) acc[i][j] = (v8f){0,0,0,0,0,0,0,0};

    const int ka = lh * 16;                 // A: 16 consecutive K at {0,16}
    const int kb = lh * 16;                 // B: 16 consecutive K at {0,16}

    const unsigned short* Ar0 = A + (size_t)(m0 + lr)       * lda;
    const unsigned short* Ar1 = A + (size_t)(m0 + 16 + lr)  * lda;
    const unsigned short* Wr0 = W + (size_t)(n0 +  0 + lr) * K;
    const unsigned short* Wr1 = W + (size_t)(n0 + 16 + lr) * K;
    const unsigned short* Wr2 = W + (size_t)(n0 + 32 + lr) * K;
    const unsigned short* Wr3 = W + (size_t)(n0 + 48 + lr) * K;

    for (int k = 0; k < K; k += 32) {
        union { v16bf v; uint4 q[2]; } a0, a1, b0, b1, b2, b3;
        a0.q[0] = *(const uint4*)(Ar0 + k + ka);
        a0.q[1] = *(const uint4*)(Ar0 + k + ka + 8);
        a1.q[0] = *(const uint4*)(Ar1 + k + ka);
        a1.q[1] = *(const uint4*)(Ar1 + k + ka + 8);
        b0.q[0] = *(const uint4*)(Wr0 + k + kb);
        b0.q[1] = *(const uint4*)(Wr0 + k + kb + 8);
        b1.q[0] = *(const uint4*)(Wr1 + k + kb);
        b1.q[1] = *(const uint4*)(Wr1 + k + kb + 8);
        b2.q[0] = *(const uint4*)(Wr2 + k + kb);
        b2.q[1] = *(const uint4*)(Wr2 + k + kb + 8);
        b3.q[0] = *(const uint4*)(Wr3 + k + kb);
        b3.q[1] = *(const uint4*)(Wr3 + k + kb + 8);

        acc[0][0] = __builtin_amdgcn_wmma_f32_16x16x32_bf16(false, a0.v, false, b0.v, (short)0, acc[0][0], false, false);
        acc[0][1] = __builtin_amdgcn_wmma_f32_16x16x32_bf16(false, a0.v, false, b1.v, (short)0, acc[0][1], false, false);
        acc[0][2] = __builtin_amdgcn_wmma_f32_16x16x32_bf16(false, a0.v, false, b2.v, (short)0, acc[0][2], false, false);
        acc[0][3] = __builtin_amdgcn_wmma_f32_16x16x32_bf16(false, a0.v, false, b3.v, (short)0, acc[0][3], false, false);
        acc[1][0] = __builtin_amdgcn_wmma_f32_16x16x32_bf16(false, a1.v, false, b0.v, (short)0, acc[1][0], false, false);
        acc[1][1] = __builtin_amdgcn_wmma_f32_16x16x32_bf16(false, a1.v, false, b1.v, (short)0, acc[1][1], false, false);
        acc[1][2] = __builtin_amdgcn_wmma_f32_16x16x32_bf16(false, a1.v, false, b2.v, (short)0, acc[1][2], false, false);
        acc[1][3] = __builtin_amdgcn_wmma_f32_16x16x32_bf16(false, a1.v, false, b3.v, (short)0, acc[1][3], false, false);
    }

#pragma unroll
    for (int mt = 0; mt < 2; ++mt)
#pragma unroll
        for (int nt = 0; nt < 4; ++nt) {
            int col = n0 + nt * 16 + lr;
            if (col < Nstore) {
                float bv = bias ? bias[col] : 0.0f;
#pragma unroll
                for (int r = 0; r < 8; ++r) {
                    int row = m0 + mt * 16 + r + lh * 8;
                    float v = acc[mt][nt][r] + bv;
                    if (out_bf16)
                        ((unsigned short*)Cout)[(size_t)row * ldc + col] = f2bf(v);
                    else
                        ((float*)Cout)[(size_t)row * ldc + col] = v;
                }
            }
        }
}

// ---------- attention: e, softmax, context (one block per batch row) ----------
__global__ __launch_bounds__(256)
void attn_kernel(const unsigned short* __restrict__ Hp,   // [B,T,H] bf16
                 const float* __restrict__ ph,             // [B,H]
                 const float* __restrict__ wscore,         // [H]
                 const unsigned short* __restrict__ bH,    // [B,T,DIN] bf16
                 unsigned short* __restrict__ x) {         // [B,1280], cols 0..511
    __shared__ float s_alpha[32];
    __shared__ float s_e[32];
    const int b    = blockIdx.x;
    const int wave = threadIdx.x >> 5;
    const int lane = threadIdx.x & 31;
    const float* phb = ph + (size_t)b * H_;

    for (int t = wave; t < T_; t += 8) {
        const unsigned short* hp = Hp + ((size_t)b * T_ + t) * H_;
        float sum = 0.0f;
        for (int j = lane; j < H_; j += 32)
            sum += tanhf(bf2f(hp[j]) + phb[j]) * wscore[j];
        for (int off = 16; off; off >>= 1) sum += __shfl_xor(sum, off, 32);
        if (lane == 0) s_e[t] = sum;
    }
    __syncthreads();
    if (wave == 0) {
        float e = (lane < T_) ? s_e[lane] : -3.0e38f;
        float m = e;
        for (int off = 16; off; off >>= 1) m = fmaxf(m, __shfl_xor(m, off, 32));
        float p = (lane < T_) ? __expf(e - m) : 0.0f;
        float s = p;
        for (int off = 16; off; off >>= 1) s += __shfl_xor(s, off, 32);
        if (lane < T_) s_alpha[lane] = p / s;
    }
    __syncthreads();
    for (int d = threadIdx.x; d < DIN_; d += 256) {
        const unsigned short* bhb = bH + (size_t)b * T_ * DIN_ + d;
        float c = 0.0f;
#pragma unroll
        for (int t = 0; t < T_; ++t) c += s_alpha[t] * bf2f(bhb[(size_t)t * DIN_]);
        x[(size_t)b * 1280 + d] = f2bf(c);
    }
}

// ---------- embedding gather into x[:,512..767] ----------
__global__ void gather_ce_kernel(const float* __restrict__ emb,
                                 const int* __restrict__ text, int s,
                                 unsigned short* __restrict__ x) {
    int idx = blockIdx.x * blockDim.x + threadIdx.x;   // B*E
    if (idx >= B_ * E_) return;
    int b = idx >> 8, e = idx & 255;
    int ch = text[(size_t)b * S_ + s];
    x[(size_t)b * 1280 + 512 + e] = f2bf(emb[(size_t)ch * E_ + e]);
}

// ---------- LSTM pointwise: gates -> c,h ; h bf16 into x[:,768..] and hs ----------
__global__ void lstm_kernel(const float* __restrict__ gates,   // [B,4H]
                            float* __restrict__ c,             // [B,H]
                            unsigned short* __restrict__ x,    // [B,1280]
                            unsigned short* __restrict__ hs,   // [B,S,H] bf16
                            int s) {
    size_t idx = (size_t)blockIdx.x * blockDim.x + threadIdx.x;  // B*H
    if (idx >= (size_t)B_ * H_) return;
    int b = (int)(idx >> 9), j = (int)(idx & 511);
    const float* g = gates + (size_t)b * (4 * H_);
    float ig = sigmoidf_(g[j]);
    float fg = sigmoidf_(g[512 + j]);
    float gg = tanhf(g[1024 + j]);
    float og = sigmoidf_(g[1536 + j]);
    float cn = fg * c[idx] + ig * gg;
    c[idx] = cn;
    unsigned short hb = f2bf(og * tanhf(cn));
    x[(size_t)b * 1280 + 768 + j] = hb;
    hs[((size_t)b * S_ + s) * H_ + j] = hb;
}

// ---------- host ----------
extern "C" void kernel_launch(void* const* d_in, const int* in_sizes, int n_in,
                              void* d_out, int out_size, void* d_ws, size_t ws_size,
                              hipStream_t stream) {
    (void)in_sizes; (void)n_in; (void)out_size; (void)ws_size;
    const float* batch_H  = (const float*)d_in[0];
    const int*   text     = (const int*)  d_in[1];
    const float* W_i2h    = (const float*)d_in[2];
    const float* W_h2h    = (const float*)d_in[3];
    const float* b_h2h    = (const float*)d_in[4];
    const float* w_score  = (const float*)d_in[5];
    const float* W_ih     = (const float*)d_in[6];
    const float* W_hh     = (const float*)d_in[7];
    const float* b_ih     = (const float*)d_in[8];
    const float* b_hh     = (const float*)d_in[9];
    const float* W_gen    = (const float*)d_in[10];
    const float* b_gen    = (const float*)d_in[11];
    const float* emb      = (const float*)d_in[12];
    float* out = (float*)d_out;

    // workspace carve-out (256B aligned)
    char* p = (char*)d_ws;
    auto carve = [&](size_t bytes) -> void* {
        void* r = (void*)p;
        p += (bytes + 255) & ~(size_t)255;
        return r;
    };
    unsigned short* bH_bf   = (unsigned short*)carve((size_t)B_ * T_ * DIN_ * 2);
    unsigned short* Hp_bf   = (unsigned short*)carve((size_t)B_ * T_ * H_ * 2);
    unsigned short* x_bf    = (unsigned short*)carve((size_t)B_ * 1280 * 2);
    unsigned short* hs_bf   = (unsigned short*)carve((size_t)B_ * S_ * H_ * 2);
    unsigned short* Wcat_bf = (unsigned short*)carve((size_t)4 * H_ * 1280 * 2);
    unsigned short* Wh2h_bf = (unsigned short*)carve((size_t)H_ * H_ * 2);
    unsigned short* Wi2h_bf = (unsigned short*)carve((size_t)H_ * DIN_ * 2);
    unsigned short* Wgen_bf = (unsigned short*)carve((size_t)256 * H_ * 2);
    float* bias_cat         = (float*)carve((size_t)4 * H_ * 4);
    float* ph               = (float*)carve((size_t)B_ * H_ * 4);
    float* gates            = (float*)carve((size_t)B_ * 4 * H_ * 4);
    float* cstate           = (float*)carve((size_t)B_ * H_ * 4);

    // ---- one-time prep ----
    {
        size_t n = (size_t)B_ * T_ * DIN_;
        cvt_bf16_kernel<<<(unsigned)((n + 255) / 256), 256, 0, stream>>>(batch_H, bH_bf, n);
        n = (size_t)H_ * DIN_;
        cvt_bf16_kernel<<<(unsigned)((n + 255) / 256), 256, 0, stream>>>(W_i2h, Wi2h_bf, n);
        n = (size_t)H_ * H_;
        cvt_bf16_kernel<<<(unsigned)((n + 255) / 256), 256, 0, stream>>>(W_h2h, Wh2h_bf, n);
        size_t nw = (size_t)4 * H_ * 1280;
        prep_wcat_kernel<<<(unsigned)((nw + 255) / 256), 256, 0, stream>>>(
            W_ih, W_hh, b_ih, b_hh, Wcat_bf, bias_cat);
        prep_wgen_kernel<<<(256 * 512) / 256, 256, 0, stream>>>(W_gen, Wgen_bf);
        size_t nz = (size_t)B_ * 1280;
        zero_init_kernel<<<(unsigned)((nz + 255) / 256), 256, 0, stream>>>(x_bf, cstate);
    }

    // ---- Hp = batch_H @ W_i2h^T  (bf16 out) : M=B*T, N=512, K=512 ----
    {
        dim3 grid((B_ * T_) / 64, 512 / 256);
        gemm_bf16_nt_kernel<<<grid, 256, 0, stream>>>(
            bH_bf, DIN_, Wi2h_bf, DIN_, (void*)Hp_bf, H_, nullptr, H_, 1);
    }

    // ---- recurrent steps ----
    for (int s = 0; s < S_; ++s) {
        // ph = h @ W_h2h^T + b_h2h : A = x[:,768..1279] (lda=1280)
        {
            dim3 grid(B_ / 64, 512 / 256);
            gemm_bf16_nt_kernel<<<grid, 256, 0, stream>>>(
                x_bf + 768, 1280, Wh2h_bf, H_, (void*)ph, H_, b_h2h, H_, 0);
        }
        // attention -> context into x[:,0..511]
        attn_kernel<<<B_, 256, 0, stream>>>(Hp_bf, ph, w_score, bH_bf, x_bf);
        // char embedding into x[:,512..767]
        gather_ce_kernel<<<(B_ * E_) / 256, 256, 0, stream>>>(emb, text, s, x_bf);
        // gates = x @ [W_ih|W_hh]^T + (b_ih+b_hh) : M=B, N=4H, K=1280
        {
            dim3 grid(B_ / 64, (4 * H_) / 256);
            gemm_bf16_nt_kernel<<<grid, 256, 0, stream>>>(
                x_bf, 1280, Wcat_bf, 1280, (void*)gates, 4 * H_, bias_cat, 4 * H_, 0);
        }
        // LSTM pointwise
        lstm_kernel<<<(B_ * H_) / 256, 256, 0, stream>>>(gates, cstate, x_bf, hs_bf, s);
    }

    // ---- probs = hs @ W_gen^T + b_gen : M=B*S, N=256(pad), store N<97 ----
    {
        dim3 grid((B_ * S_) / 64, 256 / 256);
        gemm_bf16_nt_kernel<<<grid, 256, 0, stream>>>(
            hs_bf, H_, Wgen_bf, H_, (void*)out, C_, b_gen, C_, 0);
    }
}
